// CCA_61993557950898
// MI455X (gfx1250) — matrix operand order repaired and checked by reading, not compiled, not run on hardware
//
#include <hip/hip_runtime.h>
#include <hip/hip_bf16.h>

typedef _Float16 v16h __attribute__((ext_vector_type(16)));
typedef float    v8f  __attribute__((ext_vector_type(8)));

#define B_   128
#define C_   32
#define H_   64
#define W_   64
#define L_   4096   // H*W
#define M_   64     // memory slots
#define NB_  64     // 8x8 local blocks

__device__ __forceinline__ v8f wmma_f16(v16h a, v16h b, v8f c) {
    // D = A(16x32) x B(32x16) + C, fp32 accumulate
    return __builtin_amdgcn_wmma_f32_16x16x32_f16(false, a, false, b, (short)0, c, false, false);
}

__device__ __forceinline__ v8f splat8(float x) {
    v8f r;
#pragma unroll
    for (int i = 0; i < 8; ++i) r[i] = x;
    return r;
}

// ---- WMMA operand gathers (wave32; this code's contiguous k = 16*half + s, same for A and B) ----
// A 16x32 f16: lane<16 -> row=lane, K 0..15; lane>=16 -> row=lane-16, K 16..31.
// element (row,k) = src[row*rs + k*ks]
__device__ __forceinline__ v16h load_A_f16(const _Float16* src, int rs, int ks, int lane) {
    const int lh = lane >> 4, lr = lane & 15;
    v16h a;
#pragma unroll
    for (int v = 0; v < 8; ++v) {
        const int kb = 2 * v + 16 * lh;
        a[2 * v]     = src[lr * rs + kb * ks];
        a[2 * v + 1] = src[lr * rs + (kb + 1) * ks];
    }
    return a;
}
__device__ __forceinline__ v16h load_A_f32(const float* src, int rs, int ks, int lane) {
    const int lh = lane >> 4, lr = lane & 15;
    v16h a;
#pragma unroll
    for (int v = 0; v < 8; ++v) {
        const int kb = 2 * v + 16 * lh;
        a[2 * v]     = (_Float16)src[lr * rs + kb * ks];
        a[2 * v + 1] = (_Float16)src[lr * rs + (kb + 1) * ks];
    }
    return a;
}
// B 32x16 where B[k][n] = src[n*rs + k]  (i.e. B = srcT, src row-major [n][k])
__device__ __forceinline__ v16h load_Bt_f16(const _Float16* src, int rs, int lane) {
    const int lh = lane >> 4, lr = lane & 15;
    v16h b;
#pragma unroll
    for (int v = 0; v < 8; ++v) {
        const int k = 16 * lh + 2 * v;
        b[2 * v]     = src[lr * rs + k];
        b[2 * v + 1] = src[lr * rs + k + 1];
    }
    return b;
}
__device__ __forceinline__ v16h load_Bt_f32(const float* src, int rs, int lane) {
    const int lh = lane >> 4, lr = lane & 15;
    v16h b;
#pragma unroll
    for (int v = 0; v < 8; ++v) {
        const int k = 16 * lh + 2 * v;
        b[2 * v]     = (_Float16)src[lr * rs + k];
        b[2 * v + 1] = (_Float16)src[lr * rs + k + 1];
    }
    return b;
}
// B 32x16 where B[k][n] = src[k*rs + n]  (src row-major K x N)
__device__ __forceinline__ v16h load_B_f16(const _Float16* src, int rs, int lane) {
    const int lh = lane >> 4, lr = lane & 15;
    v16h b;
#pragma unroll
    for (int v = 0; v < 8; ++v) {
        const int k = 16 * lh + 2 * v;
        b[2 * v]     = src[k * rs + lr];
        b[2 * v + 1] = src[(k + 1) * rs + lr];
    }
    return b;
}
// accumulator scatter: elem r -> (row = r + 8*half, col = colbase + lane%16)
__device__ __forceinline__ void store_acc_f32(float* dst, int rs, int colbase, int lane, v8f acc) {
    const int lh = lane >> 4, lr = lane & 15;
#pragma unroll
    for (int r = 0; r < 8; ++r) dst[(r + 8 * lh) * rs + colbase + lr] = acc[r];
}
__device__ __forceinline__ void store_acc_f16(_Float16* dst, int rs, int colbase, int lane, v8f acc, float mul) {
    const int lh = lane >> 4, lr = lane & 15;
#pragma unroll
    for (int r = 0; r < 8; ++r) dst[(r + 8 * lh) * rs + colbase + lr] = (_Float16)(acc[r] * mul);
}

// ---------------- tiny precompute kernels ----------------
__global__ __launch_bounds__(256) void kv_global_kernel(
    const float* __restrict__ mem, const float* __restrict__ Wqkv,
    const float* __restrict__ bqkv, _Float16* __restrict__ kp, _Float16* __restrict__ vp) {
    for (int t = threadIdx.x; t < 2 * M_ * C_; t += 256) {
        const int d = t & 31, m = (t >> 5) & 63, kv = t >> 11;
        const float* Wrow = Wqkv + (C_ * (kv + 1) + d) * C_;
        const float* mr = mem + m * C_;
        float s = bqkv[C_ * (kv + 1) + d];
#pragma unroll
        for (int c = 0; c < C_; ++c) s += mr[c] * Wrow[c];
        (kv ? vp : kp)[m * C_ + d] = (_Float16)s;
    }
}

__global__ __launch_bounds__(256) void kv_local_kernel(
    const float* __restrict__ bmem, const float* __restrict__ Wqkv,
    const float* __restrict__ bqkv, _Float16* __restrict__ kp, _Float16* __restrict__ vp) {
    const int n = blockIdx.x;
    for (int t = threadIdx.x; t < 2 * M_ * C_; t += 256) {
        const int d = t & 31, m = (t >> 5) & 63, kv = t >> 11;
        const float* Wrow = Wqkv + (C_ * (kv + 1) + d) * C_;
        const float* br = bmem + ((size_t)n * M_ + m) * C_;
        float s = bqkv[C_ * (kv + 1) + d];
#pragma unroll
        for (int c = 0; c < C_; ++c) s += br[c] * Wrow[c];
        (kv ? vp : kp)[((size_t)n * M_ + m) * C_ + d] = (_Float16)s;
    }
}

__global__ __launch_bounds__(256) void pool_kernel(
    const float* __restrict__ feat, float* __restrict__ pooled) {
    const int idx = blockIdx.x * 256 + threadIdx.x;   // (b*64 + n)*32 + c
    if (idx >= B_ * NB_ * C_) return;
    const int c = idx & 31, n = (idx >> 5) & 63, b = idx >> 11;
    const int y0 = (n >> 3) * 8, x0 = (n & 7) * 8;
    const float* base = feat + ((size_t)b * C_ + c) * L_;
    float s = 0.f;
#pragma unroll
    for (int dy = 0; dy < 8; ++dy)
#pragma unroll
        for (int dx = 0; dx < 8; ++dx) s += base[(y0 + dy) * W_ + x0 + dx];
    pooled[idx] = s * (1.f / 64.f);
}

// ---------------- local per-block attention (1 wave = 16 batches x 1 block) ----------------
__global__ __launch_bounds__(128) void attn_local_kernel(
    const float* __restrict__ pooled, const float* __restrict__ Wqkv,
    const float* __restrict__ bqkv, const float* __restrict__ Wo,
    const float* __restrict__ bo, const _Float16* __restrict__ kp_l,
    const _Float16* __restrict__ vp_l, float* __restrict__ lo_small) {
    __shared__ float    wbuf[4][16 * M_];
    __shared__ _Float16 abuf[4][16 * M_];
    const int wv = threadIdx.x >> 5, lane = threadIdx.x & 31;
    const int lh = lane >> 4, lr = lane & 15;
    const int tile = blockIdx.x * 4 + wv;         // 512 tiles
    const int n = tile & 63, b0 = (tile >> 6) * 16;
    const float scale = 0.17677669529663687f;     // 1/sqrt(C)

    // qp = pooled_tile @ Wq^T + bq  (scaled, to f16 A staging)
    v16h aq = load_A_f32(pooled + ((size_t)b0 * NB_ + n) * C_, NB_ * C_, 1, lane);
#pragma unroll
    for (int j = 0; j < 2; ++j) {
        v16h bw = load_Bt_f32(Wqkv + (j * 16) * C_, C_, lane);
        v8f q = wmma_f16(aq, bw, splat8(bqkv[j * 16 + lr]));
        store_acc_f16(abuf[wv], C_, j * 16, lane, q, scale);
    }
    __syncthreads();

    // scores = qp @ kp^T
    v16h aq2 = load_A_f16(abuf[wv], C_, 1, lane);
    const _Float16* kpn = kp_l + (size_t)n * M_ * C_;
    const _Float16* vpn = vp_l + (size_t)n * M_ * C_;
#pragma unroll
    for (int j = 0; j < 4; ++j) {
        v16h bk = load_Bt_f16(kpn + (j * 16) * C_, C_, lane);
        v8f s = wmma_f16(aq2, bk, splat8(0.f));
        store_acc_f32(wbuf[wv], M_, j * 16, lane, s);
    }
    __syncthreads();

    if (lane < 16) {  // row-wise softmax over M=64
        float* row = &wbuf[wv][lane * M_];
        float mx = row[0];
        for (int m = 1; m < M_; ++m) mx = fmaxf(mx, row[m]);
        float sum = 0.f;
        for (int m = 0; m < M_; ++m) { float e = __expf(row[m] - mx); sum += e; row[m] = e; }
        const float inv = 1.f / sum;
        _Float16* arow = &abuf[wv][lane * M_];
        for (int m = 0; m < M_; ++m) arow[m] = (_Float16)(row[m] * inv);
    }
    __syncthreads();

    // out = a @ vp
    v16h a0 = load_A_f16(abuf[wv], M_, 1, lane);
    v16h a1 = load_A_f16(abuf[wv] + 32, M_, 1, lane);
    v8f av[2];
#pragma unroll
    for (int jd = 0; jd < 2; ++jd) {
        av[jd] = wmma_f16(a0, load_B_f16(vpn + jd * 16, C_, lane), splat8(0.f));
        av[jd] = wmma_f16(a1, load_B_f16(vpn + 32 * C_ + jd * 16, C_, lane), av[jd]);
    }
    __syncthreads();
#pragma unroll
    for (int jd = 0; jd < 2; ++jd) store_acc_f16(abuf[wv], C_, jd * 16, lane, av[jd], 1.f);
    __syncthreads();

    // proj = av @ Wo^T + bo
    v16h ao = load_A_f16(abuf[wv], C_, 1, lane);
    v8f po[2];
#pragma unroll
    for (int j = 0; j < 2; ++j) {
        v16h bwo = load_Bt_f32(Wo + (j * 16) * C_, C_, lane);
        po[j] = wmma_f16(ao, bwo, splat8(bo[j * 16 + lr]));
    }

    // scatter to lo_small[B][C][8][8]
    const int y = n >> 3, x = n & 7;
#pragma unroll
    for (int j = 0; j < 2; ++j)
#pragma unroll
        for (int r = 0; r < 8; ++r) {
            const int row = r + 8 * lh, c = j * 16 + lr;
            lo_small[(((size_t)(b0 + row) * C_ + c) * 8 + y) * 8 + x] = po[j][r];
        }
}

// ---------------- global attention + LayerNorm + fused epilogue ----------------
__global__ __launch_bounds__(256) void attn_global_kernel(
    const float* __restrict__ feat, const float* __restrict__ Wqkv,
    const float* __restrict__ bqkv, const float* __restrict__ Wo,
    const float* __restrict__ bo, const float* __restrict__ ln_g,
    const float* __restrict__ ln_b, const _Float16* __restrict__ kp,
    const _Float16* __restrict__ vp, const float* __restrict__ lo_small,
    float* __restrict__ out) {
    __shared__ _Float16 fth[C_ * 128];         // feat tile [c][px], f16
    __shared__ float    wbuf[8][16 * M_];      // per-wave f32 scratch
    __shared__ _Float16 abuf[8][16 * M_];      // per-wave f16 A staging
    __shared__ float    mu_s[8][16];
    __shared__ float    rs_s[8][16];

    const int b = blockIdx.x >> 5;
    const int p0 = (blockIdx.x & 31) * 128;
    const int wv = threadIdx.x >> 5, lane = threadIdx.x & 31;
    const int lh = lane >> 4, lr = lane & 15;
    const float scale = 0.17677669529663687f;

    for (int i = threadIdx.x; i < C_ * 128; i += 256)
        fth[i] = (_Float16)feat[((size_t)b * C_ + (i >> 7)) * L_ + p0 + (i & 127)];
    __syncthreads();

    // qp = feat_tile @ Wq^T + bq  (scaled)
    v16h aq = load_A_f16(&fth[wv * 16], 1, 128, lane);
#pragma unroll
    for (int j = 0; j < 2; ++j) {
        v16h bw = load_Bt_f32(Wqkv + (j * 16) * C_, C_, lane);
        v8f q = wmma_f16(aq, bw, splat8(bqkv[j * 16 + lr]));
        store_acc_f16(abuf[wv], C_, j * 16, lane, q, scale);
    }
    __syncthreads();

    // scores = qp @ kp^T
    v16h aq2 = load_A_f16(abuf[wv], C_, 1, lane);
#pragma unroll
    for (int j = 0; j < 4; ++j) {
        v16h bk = load_Bt_f16(kp + (j * 16) * C_, C_, lane);
        v8f s = wmma_f16(aq2, bk, splat8(0.f));
        store_acc_f32(wbuf[wv], M_, j * 16, lane, s);
    }
    __syncthreads();

    if (lane < 16) {  // softmax over M=64
        float* row = &wbuf[wv][lane * M_];
        float mx = row[0];
        for (int m = 1; m < M_; ++m) mx = fmaxf(mx, row[m]);
        float sum = 0.f;
        for (int m = 0; m < M_; ++m) { float e = __expf(row[m] - mx); sum += e; row[m] = e; }
        const float inv = 1.f / sum;
        _Float16* arow = &abuf[wv][lane * M_];
        for (int m = 0; m < M_; ++m) arow[m] = (_Float16)(row[m] * inv);
    }
    __syncthreads();

    // a @ vp
    v16h a0 = load_A_f16(abuf[wv], M_, 1, lane);
    v16h a1 = load_A_f16(abuf[wv] + 32, M_, 1, lane);
    v8f av[2];
#pragma unroll
    for (int jd = 0; jd < 2; ++jd) {
        av[jd] = wmma_f16(a0, load_B_f16(vp + jd * 16, C_, lane), splat8(0.f));
        av[jd] = wmma_f16(a1, load_B_f16(vp + 32 * C_ + jd * 16, C_, lane), av[jd]);
    }
    __syncthreads();
#pragma unroll
    for (int jd = 0; jd < 2; ++jd) store_acc_f16(abuf[wv], C_, jd * 16, lane, av[jd], 1.f);
    __syncthreads();

    // proj @ Wo^T + bo -> wbuf [row][c]
    v16h ao = load_A_f16(abuf[wv], C_, 1, lane);
#pragma unroll
    for (int j = 0; j < 2; ++j) {
        v16h bwo = load_Bt_f32(Wo + (j * 16) * C_, C_, lane);
        v8f p = wmma_f16(ao, bwo, splat8(bo[j * 16 + lr]));
        store_acc_f32(wbuf[wv], C_, j * 16, lane, p);
    }
    __syncthreads();

    if (lane < 16) {  // LayerNorm stats per pixel-row
        const float* row = &wbuf[wv][lane * C_];
        float mu = 0.f;
        for (int c = 0; c < C_; ++c) mu += row[c];
        mu *= (1.f / 32.f);
        float var = 0.f;
        for (int c = 0; c < C_; ++c) { float d = row[c] - mu; var += d * d; }
        var *= (1.f / 32.f);
        mu_s[wv][lane] = mu;
        rs_s[wv][lane] = rsqrtf(var + 1e-5f);
    }
    __syncthreads();

    // fused epilogue: LN + feat + bilinear(lo_small), coalesced stores
    for (int i = threadIdx.x; i < C_ * 128; i += 256) {
        const int c = i >> 7, px = i & 127;
        const int w2 = px >> 4, r2 = px & 15;
        const float x = wbuf[w2][r2 * C_ + c];
        const float g = (x - mu_s[w2][r2]) * rs_s[w2][r2] * ln_g[c] + ln_b[c];
        const int p = p0 + px, h = p >> 6, w = p & 63;
        const float sy = (h + 0.5f) * 0.125f - 0.5f;
        const float sx = (w + 0.5f) * 0.125f - 0.5f;
        const float fy0 = floorf(sy), fx0 = floorf(sx);
        const float fy = sy - fy0, fx = sx - fx0;
        int y0 = (int)fy0, x0 = (int)fx0;
        const int y0c = y0 < 0 ? 0 : (y0 > 7 ? 7 : y0);
        const int y1c = (y0 + 1) < 0 ? 0 : ((y0 + 1) > 7 ? 7 : y0 + 1);
        const int x0c = x0 < 0 ? 0 : (x0 > 7 ? 7 : x0);
        const int x1c = (x0 + 1) < 0 ? 0 : ((x0 + 1) > 7 ? 7 : x0 + 1);
        const float* lob = lo_small + ((size_t)b * C_ + c) * 64;
        const float v00 = lob[y0c * 8 + x0c], v01 = lob[y0c * 8 + x1c];
        const float v10 = lob[y1c * 8 + x0c], v11 = lob[y1c * 8 + x1c];
        const float lo = (1.f - fy) * ((1.f - fx) * v00 + fx * v01)
                       + fy * ((1.f - fx) * v10 + fx * v11);
        const size_t gi = ((size_t)b * C_ + c) * L_ + p;
        out[gi] = g + feat[gi] + lo;
    }
}

extern "C" void kernel_launch(void* const* d_in, const int* in_sizes, int n_in,
                              void* d_out, int out_size, void* d_ws, size_t ws_size,
                              hipStream_t stream) {
    (void)in_sizes; (void)n_in; (void)out_size; (void)ws_size;
    const float* feat   = (const float*)d_in[0];
    const float* mem    = (const float*)d_in[1];
    const float* bmem   = (const float*)d_in[2];
    const float* Wqkv_g = (const float*)d_in[3];
    const float* bqkv_g = (const float*)d_in[4];
    const float* Wo_g   = (const float*)d_in[5];
    const float* bo_g   = (const float*)d_in[6];
    const float* Wqkv_l = (const float*)d_in[7];
    const float* bqkv_l = (const float*)d_in[8];
    const float* Wo_l   = (const float*)d_in[9];
    const float* bo_l   = (const float*)d_in[10];
    const float* ln_g   = (const float*)d_in[11];
    const float* ln_b   = (const float*)d_in[12];
    float* out = (float*)d_out;

    char* ws = (char*)d_ws;
    _Float16* kp_g = (_Float16*)ws;  ws += (size_t)M_ * C_ * 2;
    _Float16* vp_g = (_Float16*)ws;  ws += (size_t)M_ * C_ * 2;
    _Float16* kp_l = (_Float16*)ws;  ws += (size_t)NB_ * M_ * C_ * 2;
    _Float16* vp_l = (_Float16*)ws;  ws += (size_t)NB_ * M_ * C_ * 2;
    float* pooled   = (float*)ws;    ws += (size_t)B_ * NB_ * C_ * 4;
    float* lo_small = (float*)ws;    ws += (size_t)B_ * C_ * 64 * 4;

    kv_global_kernel<<<1, 256, 0, stream>>>(mem, Wqkv_g, bqkv_g, kp_g, vp_g);
    kv_local_kernel<<<NB_, 256, 0, stream>>>(bmem, Wqkv_l, bqkv_l, kp_l, vp_l);
    pool_kernel<<<(B_ * NB_ * C_) / 256, 256, 0, stream>>>(feat, pooled);
    attn_local_kernel<<<128, 128, 0, stream>>>(pooled, Wqkv_l, bqkv_l, Wo_l, bo_l,
                                               kp_l, vp_l, lo_small);
    attn_global_kernel<<<B_ * 32, 256, 0, stream>>>(feat, Wqkv_g, bqkv_g, Wo_g, bo_g,
                                                    ln_g, ln_b, kp_g, vp_g, lo_small, out);
}
